// MyPVT_18021682774439
// MI455X (gfx1250) — matrix operand order repaired and checked by reading, not compiled, not run on hardware
//
#include <hip/hip_runtime.h>
#include <hip/hip_bf16.h>
#include <math.h>
#include <stdint.h>

// ---------------------------------------------------------------------------
// MI455X (gfx1250) PVT-style token block.
// B=8 N=4096 N0=Ns=16384 C=128 H=W=64 hid=512 heads=2 d=64 SR=4.
// - All dense GEMMs: v_wmma_f32_16x16x32_bf16, fp32 accumulate.
//   A-operands produced in bf16; weights pre-transposed to bf16 [n][k].
//   Global->LDS tile staging uses GLOBAL_LOAD_ASYNC_TO_LDS_B128 (ASYNCcnt).
// - Fused attention: 1 wave / 16 queries, S=QK^T (WMMA), softmax via wave32
//   shuffles, O=PV (WMMA) with V^T staged in LDS.
// - token2map / map2token via f32 atomics.
// Workspace: ~300 MiB of d_ws.
// ---------------------------------------------------------------------------

typedef __attribute__((ext_vector_type(16))) __bf16 v16bf;
typedef __attribute__((ext_vector_type(8)))  float  v8f;

// A-fragment K base for VGPR-pair p (0..7), lane group g: elements K, K+1
__device__ __forceinline__ int kbaseA(int p, int g) {
    return g * 16 + p * 2;
}

// Per-lane async copy of 16 bytes global -> LDS (ASYNCcnt-tracked).
__device__ __forceinline__ void async_copy_b128(uint32_t lds_off, const void* gptr) {
    asm volatile("global_load_async_to_lds_b128 %0, %1, off"
                 :: "v"(lds_off), "v"((uint64_t)(uintptr_t)gptr)
                 : "memory");
}
__device__ __forceinline__ void async_wait0() {
    asm volatile("s_wait_asynccnt 0x0" ::: "memory");
}

// ---------------------------------------------------------------------------
// LayerNorm over rows of 128: one wave per row, 8 rows/block; templated output.
// ---------------------------------------------------------------------------
template <typename OutT>
__global__ __launch_bounds__(256)
void ln_rows_t(const float* __restrict__ X, const float* __restrict__ gw,
               const float* __restrict__ bw, OutT* __restrict__ Y, int rows)
{
    int row  = blockIdx.x * 8 + (threadIdx.x >> 5);
    int lane = threadIdx.x & 31;
    if (row >= rows) return;
    const float* xr = X + (size_t)row * 128;
    float v[4]; float s = 0.f;
    #pragma unroll
    for (int i = 0; i < 4; i++) { v[i] = xr[lane + 32 * i]; s += v[i]; }
    #pragma unroll
    for (int m = 16; m >= 1; m >>= 1) s += __shfl_xor(s, m, 32);
    float mean = s * (1.f / 128.f);
    float s2 = 0.f;
    #pragma unroll
    for (int i = 0; i < 4; i++) { float d = v[i] - mean; s2 += d * d; }
    #pragma unroll
    for (int m = 16; m >= 1; m >>= 1) s2 += __shfl_xor(s2, m, 32);
    float rstd = rsqrtf(s2 * (1.f / 128.f) + 1e-5f);
    OutT* yr = Y + (size_t)row * 128;
    #pragma unroll
    for (int i = 0; i < 4; i++) {
        int c = lane + 32 * i;
        yr[c] = (OutT)((v[i] - mean) * rstd * gw[c] + bw[c]);
    }
}

// transpose + bf16-convert weights: wt[n*K+k] = (bf16)w[k*N+n]
__global__ void transp_w_bf16(const float* __restrict__ w, __bf16* __restrict__ wt,
                              int K, int N)
{
    int i = blockIdx.x * 256 + threadIdx.x;
    if (i >= K * N) return;
    int n = i / K, k = i - n * K;
    wt[i] = (__bf16)w[(size_t)k * N + n];
}

// ---------------------------------------------------------------------------
// GEMM: Out[M,N](f32) = A[M,K](bf16) * Bt[N,K](bf16, pre-transposed)
//                       (+bias[N]) (+resid[M,N]).
// 128 threads = 4 waves; block tile 128x64; wave owns 32 rows ->
// 2 A-frags x 4 B-frags = 8 WMMAs per 32-wide k-step.
// Tiles staged with GLOBAL_LOAD_ASYNC_TO_LDS_B128.
// Requires M%128==0, N%64==0, K%32==0 (true at all call sites).
// ---------------------------------------------------------------------------
#define GS 48   // LDS row stride (bf16 elems) = 96B, 16B-aligned for async B128
__global__ __launch_bounds__(128)
void gemm_bf16_wmma(const __bf16* __restrict__ A, const __bf16* __restrict__ Bt,
                    const float* __restrict__ bias, const float* __restrict__ resid,
                    float* __restrict__ Out, int M, int N, int K)
{
    __shared__ __bf16 As[128 * GS];
    __shared__ __bf16 Bs[64 * GS];
    const int tid = threadIdx.x, lane = tid & 31, wave = tid >> 5;
    const int g = lane >> 4, lm = lane & 15;
    const int m0 = blockIdx.x * 128, n0 = blockIdx.y * 64;
    const uint32_t asBase = (uint32_t)(uintptr_t)(&As[0]);
    const uint32_t bsBase = (uint32_t)(uintptr_t)(&Bs[0]);

    v8f acc[2][4] = {};

    for (int k0 = 0; k0 < K; k0 += 32) {
        // A tile: 128 rows x 32 k = 512 x 16B chunks
        #pragma unroll
        for (int t = 0; t < 4; t++) {
            int l = tid + t * 128;
            int row = l >> 2, ch = l & 3;
            async_copy_b128(asBase + (uint32_t)(row * GS + ch * 8) * 2,
                            A + (size_t)(m0 + row) * K + k0 + ch * 8);
        }
        // B tile (already [n][k]): 64 rows x 32 k = 256 x 16B chunks
        #pragma unroll
        for (int t = 0; t < 2; t++) {
            int l = tid + t * 128;
            int n = l >> 2, ch = l & 3;
            async_copy_b128(bsBase + (uint32_t)(n * GS + ch * 8) * 2,
                            Bt + (size_t)(n0 + n) * K + k0 + ch * 8);
        }
        async_wait0();
        __syncthreads();

        v16bf bfr[4];
        #pragma unroll
        for (int nt = 0; nt < 4; nt++) {
            const __bf16* bp = &Bs[(nt * 16 + lm) * GS + g * 16];  // K = 16g+i
            #pragma unroll
            for (int i = 0; i < 16; i++) bfr[nt][i] = bp[i];
        }
        #pragma unroll
        for (int rt = 0; rt < 2; rt++) {
            v16bf af;
            const int arow = wave * 32 + rt * 16 + lm;
            #pragma unroll
            for (int p = 0; p < 8; p++) {
                int kb = kbaseA(p, g);
                af[2 * p]     = As[arow * GS + kb];
                af[2 * p + 1] = As[arow * GS + kb + 1];
            }
            #pragma unroll
            for (int nt = 0; nt < 4; nt++)
                acc[rt][nt] = __builtin_amdgcn_wmma_f32_16x16x32_bf16(
                    false, af, false, bfr[nt], (short)0, acc[rt][nt], false, false);
        }
        __syncthreads();
    }

    #pragma unroll
    for (int rt = 0; rt < 2; rt++)
        #pragma unroll
        for (int nt = 0; nt < 4; nt++) {
            int col = n0 + nt * 16 + lm;
            float bv = bias ? bias[col] : 0.f;
            #pragma unroll
            for (int r = 0; r < 8; r++) {
                int row = m0 + wave * 32 + rt * 16 + r + 8 * g;
                size_t o = (size_t)row * N + col;
                float v = acc[rt][nt][r] + bv;
                if (resid) v += resid[o];
                Out[o] = v;
            }
        }
}

// ---------------------------------------------------------------------------
__device__ __forceinline__ int grid_idx_fn(float lx, float ly) {
    lx = fminf(fmaxf(lx, -1.f), 1.f);
    ly = fminf(fmaxf(ly, -1.f), 1.f);
    int xi = (int)rintf((lx + 1.f) * 0.5f * 63.f);
    int yi = (int)rintf((ly + 1.f) * 0.5f * 63.f);
    return yi * 64 + xi;
}

// token2map scatter for tmp = concat(xsn, conf): 129 ch, 4 threads/entry
__global__ __launch_bounds__(256)
void scatter_tmp(const float* __restrict__ xsn, const float* __restrict__ conf,
                 const float* __restrict__ loc, const int* __restrict__ idx_src,
                 float* __restrict__ num, float* __restrict__ cnt)
{
    int e = blockIdx.x * 256 + threadIdx.x;   // B*N0*4
    int part = e & 3; int j = e >> 2;
    if (j >= 8 * 16384) return;
    int b  = j >> 14;
    int hw = grid_idx_fn(loc[2 * (size_t)j], loc[2 * (size_t)j + 1]);
    int src = idx_src[j];
    const float* f = xsn + ((size_t)b * 16384 + src) * 128 + part * 32;
    float* dst = num + ((size_t)b * 4096 + hw) * 129 + part * 32;
    for (int c = 0; c < 32; c++) atomicAdd(&dst[c], f[c]);
    if (part == 3)
        atomicAdd(&num[((size_t)b * 4096 + hw) * 129 + 128],
                  conf[(size_t)b * 16384 + src]);
    if (part == 0) atomicAdd(&cnt[b * 4096 + hw], 1.f);
}

// 4x4/s4 VALID conv over finalized 64x64x128 map -> (B,256,128)
__global__ __launch_bounds__(128)
void sr_conv(const float* __restrict__ num, const float* __restrict__ cnt,
             const float* __restrict__ W4, const float* __restrict__ bias,
             float* __restrict__ out)
{
    __shared__ float patch[16 * 128];
    int blk = blockIdx.x;                 // b*256 + oy*16 + ox
    int b = blk >> 8, oy = (blk >> 4) & 15, ox = blk & 15;
    int t = threadIdx.x;                  // co
    for (int p = 0; p < 16; p++) {
        int hw = (oy * 4 + (p >> 2)) * 64 + ox * 4 + (p & 3);
        float inv = 1.f / fmaxf(cnt[b * 4096 + hw], 1.f);
        patch[p * 128 + t] = num[((size_t)b * 4096 + hw) * 129 + t] * inv;
    }
    __syncthreads();
    float acc = bias[t];
    for (int p = 0; p < 16; p++) {
        const float* wp = W4 + (size_t)p * 128 * 128 + t;   // [p][ci][co]
        const float* pp = patch + p * 128;
        #pragma unroll 4
        for (int ci = 0; ci < 128; ci++) acc += pp[ci] * wp[(size_t)ci * 128];
    }
    out[(size_t)blk * 128 + t] = acc;
}

__global__ void conf_pool(const float* __restrict__ num, const float* __restrict__ cnt,
                          float* __restrict__ confp)
{
    int i = blockIdx.x * 256 + threadIdx.x;  // B*256
    if (i >= 8 * 256) return;
    int b = i >> 8, oy = (i >> 4) & 15, ox = i & 15;
    float s = 0.f;
    for (int p = 0; p < 16; p++) {
        int hw = (oy * 4 + (p >> 2)) * 64 + ox * 4 + (p & 3);
        s += num[((size_t)b * 4096 + hw) * 129 + 128] / fmaxf(cnt[b * 4096 + hw], 1.f);
    }
    confp[i] = s * (1.f / 16.f);
}

// ---------------------------------------------------------------------------
// Fused attention (bf16 output so proj GEMM can async-load it).
// ---------------------------------------------------------------------------
__global__ __launch_bounds__(32)
void attention_fused(const float* __restrict__ q, const float* __restrict__ kv,
                     const float* __restrict__ confp, __bf16* __restrict__ out)
{
    __shared__ __bf16 Plds[16 * 264];
    __shared__ __bf16 Vt[64 * 264];          // [dd][kk]
    int blk = blockIdx.x;                     // b*512 + h*256 + qt
    int qt = blk & 255, h = (blk >> 8) & 1, b = blk >> 9;
    int lane = threadIdx.x, g = lane >> 4, lm = lane & 15;
    const float* kvb = kv + (size_t)b * 256 * 256;

    for (int l = lane; l < 64 * 256; l += 32) {
        int kk = l >> 6, dd = l & 63;
        Vt[dd * 264 + kk] = (__bf16)kvb[(size_t)kk * 256 + 128 + h * 64 + dd];
    }

    // ---- S = Q K^T ----
    v8f acc[16] = {};
    const float* qb = q + ((size_t)b * 4096 + (size_t)qt * 16) * 128 + h * 64;
    for (int ks = 0; ks < 2; ks++) {
        v16bf af;
        const float* qr = qb + (size_t)lm * 128 + ks * 32;
        #pragma unroll
        for (int p = 0; p < 8; p++) {
            int kb = kbaseA(p, g);
            af[2 * p] = (__bf16)qr[kb]; af[2 * p + 1] = (__bf16)qr[kb + 1];
        }
        #pragma unroll
        for (int nt = 0; nt < 16; nt++) {
            v16bf bk;   // col = key nt*16+lm, element i -> K = ks*32+16g+i
            const float* kr = kvb + (size_t)(nt * 16 + lm) * 256 + h * 64 + ks * 32 + g * 16;
            #pragma unroll
            for (int i = 0; i < 16; i++) bk[i] = (__bf16)kr[i];
            acc[nt] = __builtin_amdgcn_wmma_f32_16x16x32_bf16(
                false, af, false, bk, (short)0, acc[nt], false, false);
        }
    }

    const float scale = 0.125f;  // 64^-0.5
    float rmax[8], rsum[8];
    #pragma unroll
    for (int r = 0; r < 8; r++) rmax[r] = -3.4e38f;
    #pragma unroll
    for (int nt = 0; nt < 16; nt++) {
        float cb = confp[b * 256 + nt * 16 + lm];
        #pragma unroll
        for (int r = 0; r < 8; r++) {
            float s = acc[nt][r] * scale + cb;
            acc[nt][r] = s;
            rmax[r] = fmaxf(rmax[r], s);
        }
    }
    #pragma unroll
    for (int r = 0; r < 8; r++)
        for (int m = 1; m < 16; m <<= 1) rmax[r] = fmaxf(rmax[r], __shfl_xor(rmax[r], m, 32));
    #pragma unroll
    for (int r = 0; r < 8; r++) rsum[r] = 0.f;
    #pragma unroll
    for (int nt = 0; nt < 16; nt++)
        #pragma unroll
        for (int r = 0; r < 8; r++) {
            float p = __expf(acc[nt][r] - rmax[r]);
            acc[nt][r] = p; rsum[r] += p;
        }
    #pragma unroll
    for (int r = 0; r < 8; r++)
        for (int m = 1; m < 16; m <<= 1) rsum[r] += __shfl_xor(rsum[r], m, 32);

    #pragma unroll
    for (int nt = 0; nt < 16; nt++)
        #pragma unroll
        for (int r = 0; r < 8; r++)
            Plds[(r + 8 * g) * 264 + nt * 16 + lm] = (__bf16)(acc[nt][r] / rsum[r]);
    __syncthreads();

    // ---- O = P V ----
    v8f o[4] = {};
    for (int ks = 0; ks < 8; ks++) {
        v16bf ap;
        #pragma unroll
        for (int p = 0; p < 8; p++) {
            int kb = ks * 32 + kbaseA(p, g);
            ap[2 * p] = Plds[lm * 264 + kb]; ap[2 * p + 1] = Plds[lm * 264 + kb + 1];
        }
        #pragma unroll
        for (int dt = 0; dt < 4; dt++) {
            v16bf bv;
            const __bf16* vp = &Vt[(dt * 16 + lm) * 264 + ks * 32 + g * 16];
            #pragma unroll
            for (int i = 0; i < 16; i++) bv[i] = vp[i];
            o[dt] = __builtin_amdgcn_wmma_f32_16x16x32_bf16(
                false, ap, false, bv, (short)0, o[dt], false, false);
        }
    }
    #pragma unroll
    for (int dt = 0; dt < 4; dt++)
        #pragma unroll
        for (int r = 0; r < 8; r++) {
            int row = b * 4096 + qt * 16 + r + 8 * g;
            out[(size_t)row * 128 + h * 64 + dt * 16 + lm] = (__bf16)o[dt][r];
        }
}

// token2map scatter for h (512 ch, 8 threads/entry)
__global__ __launch_bounds__(256)
void scatter_h(const float* __restrict__ h, const float* __restrict__ loc,
               const int* __restrict__ idx_agg, float* __restrict__ num,
               float* __restrict__ cnt)
{
    int e = blockIdx.x * 256 + threadIdx.x;   // B*N0*8
    int part = e & 7; int j = e >> 3;
    if (j >= 8 * 16384) return;
    int b  = j >> 14;
    int hw = grid_idx_fn(loc[2 * (size_t)j], loc[2 * (size_t)j + 1]);
    int t  = idx_agg[j];
    const float* f = h + ((size_t)b * 4096 + t) * 512 + part * 64;
    float* dst = num + ((size_t)b * 4096 + hw) * 512 + part * 64;
    for (int c = 0; c < 64; c++) atomicAdd(&dst[c], f[c]);
    if (part == 0) atomicAdd(&cnt[b * 4096 + hw], 1.f);
}

__global__ void finalize_map(float* __restrict__ num, const float* __restrict__ cnt)
{
    size_t i = (size_t)blockIdx.x * 256 + threadIdx.x;   // B*4096*512
    if (i >= (size_t)8 * 4096 * 512) return;
    num[i] = num[i] / fmaxf(cnt[i >> 9], 1.f);
}

// depthwise 3x3 SAME conv on (B,64,64,512)
__global__ __launch_bounds__(256)
void dwconv3x3(const float* __restrict__ src, const float* __restrict__ w9,
               const float* __restrict__ bias, float* __restrict__ dst)
{
    int blk = blockIdx.x;                     // b*4096 + y*64 + x
    int b = blk >> 12, y = (blk >> 6) & 63, x = blk & 63;
    for (int cc = threadIdx.x; cc < 512; cc += 256) {
        float acc = bias[cc];
        #pragma unroll
        for (int dy = -1; dy <= 1; dy++) {
            int yy = y + dy; if (yy < 0 || yy > 63) continue;
            #pragma unroll
            for (int dx = -1; dx <= 1; dx++) {
                int xx = x + dx; if (xx < 0 || xx > 63) continue;
                acc += src[((size_t)b * 4096 + yy * 64 + xx) * 512 + cc]
                     * w9[((dy + 1) * 3 + (dx + 1)) * 512 + cc];
            }
        }
        dst[((size_t)b * 4096 + y * 64 + x) * 512 + cc] = acc;
    }
}

// map2token scatter (weighted)
__global__ __launch_bounds__(256)
void scatter_m2t(const float* __restrict__ hdw, const float* __restrict__ loc,
                 const int* __restrict__ idx_agg, const float* __restrict__ aggw,
                 float* __restrict__ num, float* __restrict__ den)
{
    int e = blockIdx.x * 256 + threadIdx.x;   // B*N0*8
    int part = e & 7; int j = e >> 3;
    if (j >= 8 * 16384) return;
    int b  = j >> 14;
    int hw = grid_idx_fn(loc[2 * (size_t)j], loc[2 * (size_t)j + 1]);
    int t  = idx_agg[j];
    float w = aggw[j];
    const float* v = hdw + ((size_t)b * 4096 + hw) * 512 + part * 64;
    float* dst = num + ((size_t)b * 4096 + t) * 512 + part * 64;
    for (int c = 0; c < 64; c++) atomicAdd(&dst[c], w * v[c]);
    if (part == 0) atomicAdd(&den[b * 4096 + t], w);
}

// ht = num/max(den,1e-6) + h*dwskip ; exact GELU ; emit bf16 (fc2 A-operand)
__global__ void combine_gelu(const float* __restrict__ num, const float* __restrict__ den,
                             const float* __restrict__ h, const float* __restrict__ dwskip,
                             __bf16* __restrict__ gout)
{
    size_t i = (size_t)blockIdx.x * 256 + threadIdx.x;
    if (i >= (size_t)8 * 4096 * 512) return;
    int c = (int)(i & 511);
    float v = num[i] / fmaxf(den[i >> 9], 1e-6f) + h[i] * dwskip[c];
    gout[i] = (__bf16)(0.5f * v * (1.f + erff(v * 0.70710678118654752f)));
}

// ---------------------------------------------------------------------------
extern "C" void kernel_launch(void* const* d_in, const int* in_sizes, int n_in,
                              void* d_out, int out_size, void* d_ws, size_t ws_size,
                              hipStream_t stream)
{
    (void)in_sizes; (void)n_in; (void)out_size; (void)ws_size;
    const float* x       = (const float*)d_in[0];
    const float* loc     = (const float*)d_in[1];
    const int*   idx_agg = (const int*)  d_in[2];
    const float* aggw    = (const float*)d_in[3];
    const float* xsrc    = (const float*)d_in[4];
    const int*   idx_src = (const int*)  d_in[5];
    const float* conf    = (const float*)d_in[6];
    const float* ln1g = (const float*)d_in[7],  *ln1b = (const float*)d_in[8];
    const float* ln2g = (const float*)d_in[9],  *ln2b = (const float*)d_in[10];
    const float* wq   = (const float*)d_in[11];
    const float* wkv  = (const float*)d_in[12];
    const float* wproj= (const float*)d_in[13], *bproj= (const float*)d_in[14];
    const float* srw  = (const float*)d_in[15], *srb  = (const float*)d_in[16];
    const float* srng = (const float*)d_in[17], *srnb = (const float*)d_in[18];
    const float* fc1w = (const float*)d_in[19], *fc1b = (const float*)d_in[20];
    const float* dww  = (const float*)d_in[21], *dwb  = (const float*)d_in[22];
    const float* dwskip=(const float*)d_in[23];
    const float* fc2w = (const float*)d_in[24], *fc2b = (const float*)d_in[25];
    float* out = (float*)d_out;

    // byte bump allocator (256B aligned) on d_ws
    char* base = (char*)d_ws; size_t off = 0;
    auto allocB = [&](size_t bytes) {
        void* r = base + off; off += (bytes + 255) & ~(size_t)255; return r;
    };
    const size_t MN  = (size_t)32768 * 128;     // (B*N, C)
    const size_t MH  = (size_t)32768 * 512;     // (B*N, hid)
    __bf16* xn_bf  = (__bf16*)allocB(MN * 2);
    float*  xsn_h  = (float*) allocB((size_t)131072 * 128 * 4); // xsn -> later h
    float*  qb     = (float*) allocB(MN * 4);
    float*  tnum   = (float*) allocB((size_t)8 * 4096 * 129 * 4);
    float*  tcnt   = (float*) allocB((size_t)8 * 4096 * 4);
    float*  xsraw  = (float*) allocB((size_t)8 * 256 * 128 * 4);
    __bf16* xs_bf  = (__bf16*)allocB((size_t)8 * 256 * 128 * 2);
    float*  confp  = (float*) allocB((size_t)8 * 256 * 4);
    float*  kvb    = (float*) allocB((size_t)8 * 256 * 256 * 4);
    __bf16* att_bf = (__bf16*)allocB(MN * 2);
    float*  x1     = (float*) allocB(MN * 4);
    __bf16* yb_bf  = (__bf16*)allocB(MN * 2);
    float*  hmap   = (float*) allocB(MH * 4);   // token2map(h) -> later htnum
    float*  hcnt   = (float*) allocB((size_t)8 * 4096 * 4);
    float*  hdw    = (float*) allocB(MH * 4);
    float*  htden  = (float*) allocB((size_t)8 * 4096 * 4);
    __bf16* gbuf   = (__bf16*)allocB(MH * 2);   // gelu(ht) bf16
    __bf16* wqT    = (__bf16*)allocB((size_t)128 * 128 * 2);
    __bf16* wkvT   = (__bf16*)allocB((size_t)256 * 128 * 2);
    __bf16* wprojT = (__bf16*)allocB((size_t)128 * 128 * 2);
    __bf16* fc1T   = (__bf16*)allocB((size_t)512 * 128 * 2);
    __bf16* fc2T   = (__bf16*)allocB((size_t)128 * 512 * 2);

    // 0) weights -> transposed bf16 [n][k]
    transp_w_bf16<<<64,  256, 0, stream>>>(wq,    wqT,    128, 128);
    transp_w_bf16<<<128, 256, 0, stream>>>(wkv,   wkvT,   128, 256);
    transp_w_bf16<<<64,  256, 0, stream>>>(wproj, wprojT, 128, 128);
    transp_w_bf16<<<256, 256, 0, stream>>>(fc1w,  fc1T,   128, 512);
    transp_w_bf16<<<256, 256, 0, stream>>>(fc2w,  fc2T,   512, 128);

    // 1) LayerNorms (xn in bf16 for GEMM; xsn in f32 for scatter)
    ln_rows_t<__bf16><<<4096, 256, 0, stream>>>(x,    ln1g, ln1b, xn_bf, 32768);
    ln_rows_t<float> <<<16384,256, 0, stream>>>(xsrc, ln1g, ln1b, xsn_h, 131072);

    // 2) Q = xn @ wq
    gemm_bf16_wmma<<<dim3(256, 2), 128, 0, stream>>>(xn_bf, wqT, nullptr, nullptr, qb, 32768, 128, 128);

    // 3) token2map(concat(xsn, conf))
    hipMemsetAsync(tnum, 0, (size_t)8 * 4096 * 129 * 4, stream);
    hipMemsetAsync(tcnt, 0, (size_t)8 * 4096 * 4, stream);
    scatter_tmp<<<2048, 256, 0, stream>>>(xsn_h, conf, loc, idx_src, tnum, tcnt);

    // 4) sr conv + LN(srn, bf16 out) + conf pool
    sr_conv  <<<2048, 128, 0, stream>>>(tnum, tcnt, srw, srb, xsraw);
    ln_rows_t<__bf16><<<256, 256, 0, stream>>>(xsraw, srng, srnb, xs_bf, 2048);
    conf_pool<<<8,    256, 0, stream>>>(tnum, tcnt, confp);

    // 5) KV = xs @ wkv
    gemm_bf16_wmma<<<dim3(16, 4), 128, 0, stream>>>(xs_bf, wkvT, nullptr, nullptr, kvb, 2048, 256, 128);

    // 6) fused attention -> att_bf
    attention_fused<<<4096, 32, 0, stream>>>(qb, kvb, confp, att_bf);

    // 7) x1 = x + att @ wproj + bproj
    gemm_bf16_wmma<<<dim3(256, 2), 128, 0, stream>>>(att_bf, wprojT, bproj, x, x1, 32768, 128, 128);

    // 8) y = LN2(x1) (bf16); h = y @ fc1 + fc1_b (f32, reuse xsn buffer)
    ln_rows_t<__bf16><<<4096, 256, 0, stream>>>(x1, ln2g, ln2b, yb_bf, 32768);
    gemm_bf16_wmma<<<dim3(256, 8), 128, 0, stream>>>(yb_bf, fc1T, fc1b, nullptr, xsn_h, 32768, 512, 128);

    // 9) token2map(h) -> finalize -> depthwise 3x3
    hipMemsetAsync(hmap, 0, MH * 4, stream);
    hipMemsetAsync(hcnt, 0, (size_t)8 * 4096 * 4, stream);
    scatter_h   <<<4096,  256, 0, stream>>>(xsn_h, loc, idx_agg, hmap, hcnt);
    finalize_map<<<65536, 256, 0, stream>>>(hmap, hcnt);
    dwconv3x3   <<<32768, 256, 0, stream>>>(hmap, dww, dwb, hdw);

    // 10) map2token (reuse hmap as htnum), skip-add, GELU -> bf16
    hipMemsetAsync(hmap,  0, MH * 4, stream);
    hipMemsetAsync(htden, 0, (size_t)8 * 4096 * 4, stream);
    scatter_m2t <<<4096,  256, 0, stream>>>(hdw, loc, idx_agg, aggw, hmap, htden);
    combine_gelu<<<65536, 256, 0, stream>>>(hmap, htden, xsn_h, dwskip, gbuf);

    // 11) out = x1 + gelu(ht) @ fc2 + fc2_b
    gemm_bf16_wmma<<<dim3(256, 2), 128, 0, stream>>>(gbuf, fc2T, fc2b, x1, out, 32768, 128, 512);
}
